// MultiHeadAttention_63522566308320
// MI455X (gfx1250) — matrix operand order repaired and checked by reading, not compiled, not run on hardware
//
#include <hip/hip_runtime.h>
#include <hip/hip_bf16.h>
#include <math.h>

typedef __bf16 bf16_t;
typedef __attribute__((ext_vector_type(16))) __bf16 v16bf;
typedef __attribute__((ext_vector_type(8)))  float  v8f;
typedef __attribute__((ext_vector_type(4)))  unsigned int v4u;
typedef __attribute__((ext_vector_type(8)))  unsigned int v8u;

// Problem dims (fixed by the reference)
constexpr int BB   = 2;
constexpr int SS   = 2048;
constexpr int DD   = 2048;
constexpr int HH   = 16;
constexpr int HDIM = 128;   // DD / HH

union AFrag {
    v16bf v;
    uint4 u[2];
};

// ---------------------------------------------------------------------------
// TDM: issue a 2D bf16 tile load (global -> LDS) via tensor_load_to_lds.
// Descriptor per CDNA5 ISA ch.8 (D# groups). Uniform args; wave-scalar op.
//   tdim0/tile0: contiguous dim (elements), tdim1/tile1: rows,
//   stride_elems: row stride, pad_*: LDS padding control (both 0 = disabled).
// ---------------------------------------------------------------------------
__device__ __forceinline__ void tdm_load_2d_bf16(
        const void* gptr, unsigned lds_off,
        unsigned tdim0, unsigned tdim1, unsigned tile0, unsigned tile1,
        unsigned stride_elems, unsigned pad_en, unsigned pad_ivl, unsigned pad_amt) {
    unsigned long long ga = (unsigned long long)(uintptr_t)gptr;
    v4u g0;
    g0[0] = 1u;                                                   // count=1 (valid, user)
    g0[1] = lds_off;                                              // lds_addr
    g0[2] = (unsigned)ga;                                         // global_addr[31:0]
    g0[3] = (unsigned)((ga >> 32) & 0x01ffffffu) | (2u << 30);    // addr[56:32] | type=2
    v8u g1;
    g1[0] = (1u << 16) | (pad_en << 20) | (pad_ivl << 22) | (pad_amt << 25); // data_size=2B
    g1[1] = (tdim0 & 0xffffu) << 16;                              // abar=0 | tensor_dim0.lo
    g1[2] = (tdim0 >> 16) | ((tdim1 & 0xffffu) << 16);            // dim0.hi | tensor_dim1.lo
    g1[3] = (tdim1 >> 16) | (tile0 << 16);                        // dim1.hi | tile_dim0
    g1[4] = tile1;                                                // tile_dim1 | tile_dim2=0
    g1[5] = stride_elems;                                         // tensor_dim0_stride.lo
    g1[6] = 0u;                                                   // stride.hi | dim1_stride.lo
    g1[7] = 0u;
    v4u gz = {0u, 0u, 0u, 0u};
    asm volatile("tensor_load_to_lds %0, %1, %2, %3"
                 :
                 : "s"(g0), "s"(g1), "s"(gz), "s"(gz)
                 : "memory");
}

// ---------------------------------------------------------------------------
// fp32 -> bf16 conversion (vectorized 4-wide)
// ---------------------------------------------------------------------------
__global__ void cvt_f32_to_bf16(const float* __restrict__ src,
                                bf16_t* __restrict__ dst, int n4) {
    int i = blockIdx.x * blockDim.x + threadIdx.x;
    if (i >= n4) return;
    float4 f = ((const float4*)src)[i];
    union { bf16_t h[4]; uint2 u; } o;
    o.h[0] = (bf16_t)f.x; o.h[1] = (bf16_t)f.y;
    o.h[2] = (bf16_t)f.z; o.h[3] = (bf16_t)f.w;
    ((uint2*)dst)[i] = o.u;
}

// ---------------------------------------------------------------------------
// bf16 GEMM: C[M,N] = A[M,K] * Bw[K,N] + bias[N]
// Block tile 128x128, K-step 64. 256 threads = 8 waves; wave tile 32x64.
// A tile copied by TDM (with LDS pad), B tile transposed manually.
// ---------------------------------------------------------------------------
template <typename OutT>
__global__ __launch_bounds__(256)
void gemm_bf16(const bf16_t* __restrict__ A, const bf16_t* __restrict__ Bw,
               const float* __restrict__ bias, OutT* __restrict__ C,
               int M, int N, int K) {
    constexpr int BM = 128, BN = 128, BK = 64, PAD = 8;
    __shared__ __align__(16) bf16_t As[BM][BK + PAD];   // row-major (TDM pads rows)
    __shared__ __align__(16) bf16_t Bt[BN][BK + PAD];   // transposed: [n][k]

    const int t    = threadIdx.x;
    const int wave = t >> 5;
    const int lane = t & 31;
    const int hl   = lane >> 4;   // 16-lane half
    const int ln   = lane & 15;
    const int wm   = wave >> 1;   // 0..3 (M direction)
    const int wn   = wave & 1;    // 0..1 (N direction)
    const int m0   = blockIdx.y * BM;
    const int n0   = blockIdx.x * BN;

    v8f acc[2][4];
    const v8f vzero = {0.f, 0.f, 0.f, 0.f, 0.f, 0.f, 0.f, 0.f};
#pragma unroll
    for (int mt = 0; mt < 2; ++mt)
#pragma unroll
        for (int nt = 0; nt < 4; ++nt) acc[mt][nt] = vzero;

    for (int k0 = 0; k0 < K; k0 += BK) {
        __syncthreads();
        // A tile 128x64 via TDM; pad_interval code 4 = 128B (one row of 64
        // bf16), pad_amount code 3 = 4 dwords = 8 bf16 -> matches BK+PAD.
        if (wave == 0) {
            tdm_load_2d_bf16(&A[(size_t)m0 * K + k0],
                             (unsigned)(uintptr_t)&As[0][0],
                             BK, BM, BK, BM, (unsigned)K,
                             1u, 4u, 3u);
        }
        // B tile: 64x128, store transposed into Bt[n][k] (overlaps the DMA)
#pragma unroll
        for (int i = 0; i < 4; ++i) {
            int c  = t + 256 * i;
            int kk = c >> 4;
            int nn = (c & 15) * 8;
            uint4 u = *(const uint4*)&Bw[(size_t)(k0 + kk) * N + n0 + nn];
            const bf16_t* e = (const bf16_t*)&u;
#pragma unroll
            for (int q = 0; q < 8; ++q) Bt[nn + q][kk] = e[q];
        }
        if (wave == 0) __builtin_amdgcn_s_wait_tensorcnt(0);
        __syncthreads();

#pragma unroll
        for (int kk = 0; kk < BK; kk += 32) {
            AFrag a[2], b[4];
#pragma unroll
            for (int mt = 0; mt < 2; ++mt) {
                int row = wm * 32 + mt * 16 + ln;
                a[mt].u[0] = *(const uint4*)&As[row][kk + hl * 8];
                a[mt].u[1] = *(const uint4*)&As[row][kk + 16 + hl * 8];
            }
#pragma unroll
            for (int nt = 0; nt < 4; ++nt) {
                int col = wn * 64 + nt * 16 + ln;
                b[nt].u[0] = *(const uint4*)&Bt[col][kk + hl * 8];
                b[nt].u[1] = *(const uint4*)&Bt[col][kk + 16 + hl * 8];
            }
#pragma unroll
            for (int mt = 0; mt < 2; ++mt)
#pragma unroll
                for (int nt = 0; nt < 4; ++nt)
                    acc[mt][nt] = __builtin_amdgcn_wmma_f32_16x16x32_bf16(
                        false, a[mt].v, false, b[nt].v,
                        (short)0, acc[mt][nt], false, false);
        }
    }

    // Epilogue: C layout -> lane ln = N, vgpr r -> M = r + 8*hl
#pragma unroll
    for (int mt = 0; mt < 2; ++mt)
#pragma unroll
        for (int nt = 0; nt < 4; ++nt) {
            int gn = n0 + wn * 64 + nt * 16 + ln;
            float bv = bias[gn];
#pragma unroll
            for (int r = 0; r < 8; ++r) {
                int gm  = m0 + wm * 32 + mt * 16 + r + 8 * hl;
                float v = acc[mt][nt][r] + bv;
                C[(size_t)gm * N + gn] = (OutT)v;
            }
        }
}

// ---------------------------------------------------------------------------
// Flash attention (causal, online softmax).
// Grid: (SS/64, BB*HH). Block: 128 threads = 4 waves; wave owns 16 q-rows.
// Q/K/V/Ctx stored bf16 [BB*SS, DD] row-major, head h = cols [h*HDIM, ...).
// K tile copied by TDM; V tile transposed manually.
// ---------------------------------------------------------------------------
__global__ __launch_bounds__(128)
void flash_attn(const bf16_t* __restrict__ Q, const bf16_t* __restrict__ Km,
                const bf16_t* __restrict__ Vm, bf16_t* __restrict__ Ctx) {
    constexpr int BQ = 64, BKEY = 32, PAD = 8;
    __shared__ __align__(16) bf16_t Ks[BKEY][HDIM];        // 32 x 128 row-major
    __shared__ __align__(16) bf16_t Vt[HDIM][BKEY + PAD];  // transposed V
    __shared__ __align__(16) bf16_t Ps[4][16][BKEY + PAD]; // per-wave P tile

    const int t    = threadIdx.x;
    const int wave = t >> 5;
    const int lane = t & 31;
    const int hl   = lane >> 4;
    const int ln   = lane & 15;
    const int bh   = blockIdx.y;
    const int b    = bh / HH;
    const int h    = bh % HH;
    const int qlo  = blockIdx.x * BQ;
    const int qr0  = qlo + wave * 16;

    const bf16_t* qbase = Q  + (size_t)b * SS * DD + (size_t)h * HDIM;
    const bf16_t* kbash = Km + (size_t)b * SS * DD + (size_t)h * HDIM;
    const bf16_t* vbash = Vm + (size_t)b * SS * DD + (size_t)h * HDIM;

    // Preload Q fragments (16 rows x 128 cols -> 4 K-fragments)
    AFrag qf[4];
#pragma unroll
    for (int kf = 0; kf < 4; ++kf) {
        const bf16_t* p = qbase + (size_t)(qr0 + ln) * DD + kf * 32;
        qf[kf].u[0] = *(const uint4*)(p + hl * 8);
        qf[kf].u[1] = *(const uint4*)(p + 16 + hl * 8);
    }

    const v8f vzero = {0.f, 0.f, 0.f, 0.f, 0.f, 0.f, 0.f, 0.f};
    v8f o[8];
#pragma unroll
    for (int ot = 0; ot < 8; ++ot) o[ot] = vzero;
    float mi[8], li[8];
#pragma unroll
    for (int r = 0; r < 8; ++r) { mi[r] = -INFINITY; li[r] = 0.f; }

    const float sc = 0.08838834764831845f;  // 1/sqrt(128)
    const int jend = 2 * blockIdx.x + 2;    // key blocks covering causal range

    for (int j = 0; j < jend; ++j) {
        __syncthreads();
        const bf16_t* kb = kbash + (size_t)(j * BKEY) * DD;
        const bf16_t* vb = vbash + (size_t)(j * BKEY) * DD;
        // K tile 32x128 via TDM (Ks stride is exactly HDIM, no pad needed)
        if (wave == 0) {
            tdm_load_2d_bf16(kb, (unsigned)(uintptr_t)&Ks[0][0],
                             HDIM, BKEY, HDIM, BKEY, (unsigned)DD,
                             0u, 0u, 0u);
        }
        // V tile transposed manually (overlaps the DMA)
#pragma unroll
        for (int i = 0; i < 4; ++i) {
            int c   = t + 128 * i;       // 0..511 chunks of 8 bf16
            int row = c >> 4;            // key row 0..31
            int col = (c & 15) * 8;      // hd col
            uint4 u = *(const uint4*)&vb[(size_t)row * DD + col];
            const bf16_t* e = (const bf16_t*)&u;
#pragma unroll
            for (int q = 0; q < 8; ++q) Vt[col + q][row] = e[q];
        }
        if (wave == 0) __builtin_amdgcn_s_wait_tensorcnt(0);
        __syncthreads();

        // S = Q * K^T  (16 x 32 per wave)
        v8f s[2] = {vzero, vzero};
#pragma unroll
        for (int kf = 0; kf < 4; ++kf) {
            AFrag bf[2];
#pragma unroll
            for (int nt = 0; nt < 2; ++nt) {
                int krow = nt * 16 + ln;
                bf[nt].u[0] = *(const uint4*)&Ks[krow][kf * 32 + hl * 8];
                bf[nt].u[1] = *(const uint4*)&Ks[krow][kf * 32 + 16 + hl * 8];
            }
#pragma unroll
            for (int nt = 0; nt < 2; ++nt)
                s[nt] = __builtin_amdgcn_wmma_f32_16x16x32_bf16(
                    false, qf[kf].v, false, bf[nt].v,
                    (short)0, s[nt], false, false);
        }

        // Online softmax. Row M = r + 8*hl lives in one 16-lane half, so
        // xor-shuffles with masks 8..1 reduce exactly one row.
        float pr[2][8];
#pragma unroll
        for (int r = 0; r < 8; ++r) {
            int qg  = qr0 + r + 8 * hl;
            int k0g = j * BKEY + ln;
            int k1g = j * BKEY + 16 + ln;
            float v0 = s[0][r] * sc; if (k0g > qg) v0 = -INFINITY;
            float v1 = s[1][r] * sc; if (k1g > qg) v1 = -INFINITY;
            float rmax = fmaxf(v0, v1);
#pragma unroll
            for (int msk = 8; msk >= 1; msk >>= 1)
                rmax = fmaxf(rmax, __shfl_xor(rmax, msk, 32));
            float mnew  = fmaxf(mi[r], rmax);
            float alpha = __expf(mi[r] - mnew);
            float p0 = __expf(v0 - mnew);
            float p1 = __expf(v1 - mnew);
            float rsum = p0 + p1;
#pragma unroll
            for (int msk = 8; msk >= 1; msk >>= 1)
                rsum += __shfl_xor(rsum, msk, 32);
            li[r] = li[r] * alpha + rsum;
            mi[r] = mnew;
#pragma unroll
            for (int ot = 0; ot < 8; ++ot) o[ot][r] *= alpha;
            pr[0][r] = p0;
            pr[1][r] = p1;
        }

        // Re-layout P (C format -> A format) through per-wave LDS tile
#pragma unroll
        for (int nt = 0; nt < 2; ++nt)
#pragma unroll
            for (int r = 0; r < 8; ++r)
                Ps[wave][r + 8 * hl][nt * 16 + ln] = (bf16_t)pr[nt][r];

        AFrag pf;
        pf.u[0] = *(const uint4*)&Ps[wave][ln][hl * 8];
        pf.u[1] = *(const uint4*)&Ps[wave][ln][16 + hl * 8];

        // O += P * V   (B fragment per lane = Vt row, table k order)
#pragma unroll
        for (int ot = 0; ot < 8; ++ot) {
            AFrag vf;
            int vr = ot * 16 + ln;
            vf.u[0] = *(const uint4*)&Vt[vr][hl * 8];
            vf.u[1] = *(const uint4*)&Vt[vr][16 + hl * 8];
            o[ot] = __builtin_amdgcn_wmma_f32_16x16x32_bf16(
                false, pf.v, false, vf.v, (short)0, o[ot], false, false);
        }
    }

    // Normalize and write ctx (bf16)
    bf16_t* cbase = Ctx + (size_t)b * SS * DD + (size_t)h * HDIM;
#pragma unroll
    for (int ot = 0; ot < 8; ++ot)
#pragma unroll
        for (int r = 0; r < 8; ++r) {
            int qg = qr0 + r + 8 * hl;
            float val = o[ot][r] / li[r];
            cbase[(size_t)qg * DD + ot * 16 + ln] = (bf16_t)val;
        }
}

// ---------------------------------------------------------------------------
// Host-side launch
// ---------------------------------------------------------------------------
extern "C" void kernel_launch(void* const* d_in, const int* in_sizes, int n_in,
                              void* d_out, int out_size, void* d_ws, size_t ws_size,
                              hipStream_t stream) {
    const float* x  = (const float*)d_in[0];
    const float* wq = (const float*)d_in[1];
    const float* bq = (const float*)d_in[2];
    const float* wk = (const float*)d_in[3];
    const float* bk = (const float*)d_in[4];
    const float* wv = (const float*)d_in[5];
    const float* bv = (const float*)d_in[6];
    const float* wp = (const float*)d_in[7];
    const float* bp = (const float*)d_in[8];
    float* out = (float*)d_out;

    const size_t XE = (size_t)BB * SS * DD;  // 8M elems
    const size_t WE = (size_t)DD * DD;       // 4M elems
    bf16_t* wsb  = (bf16_t*)d_ws;
    bf16_t* xb   = wsb;
    bf16_t* wqb  = xb   + XE;
    bf16_t* wkb  = wqb  + WE;
    bf16_t* wvb  = wkb  + WE;
    bf16_t* wpb  = wvb  + WE;
    bf16_t* qb   = wpb  + WE;
    bf16_t* kb   = qb   + XE;
    bf16_t* vb   = kb   + XE;
    bf16_t* ctxb = vb   + XE;

    // 1) convert to bf16
    {
        int n4 = (int)(XE / 4);
        cvt_f32_to_bf16<<<(n4 + 255) / 256, 256, 0, stream>>>(x, xb, n4);
        int w4 = (int)(WE / 4);
        cvt_f32_to_bf16<<<(w4 + 255) / 256, 256, 0, stream>>>(wq, wqb, w4);
        cvt_f32_to_bf16<<<(w4 + 255) / 256, 256, 0, stream>>>(wk, wkb, w4);
        cvt_f32_to_bf16<<<(w4 + 255) / 256, 256, 0, stream>>>(wv, wvb, w4);
        cvt_f32_to_bf16<<<(w4 + 255) / 256, 256, 0, stream>>>(wp, wpb, w4);
    }

    // 2) QKV projections
    {
        const int M = BB * SS, N = DD, K = DD;
        dim3 grid(N / 128, M / 128);
        gemm_bf16<bf16_t><<<grid, 256, 0, stream>>>(xb, wqb, bq, qb, M, N, K);
        gemm_bf16<bf16_t><<<grid, 256, 0, stream>>>(xb, wkb, bk, kb, M, N, K);
        gemm_bf16<bf16_t><<<grid, 256, 0, stream>>>(xb, wvb, bv, vb, M, N, K);
    }

    // 3) causal flash attention
    {
        dim3 grid(SS / 64, BB * HH);
        flash_attn<<<grid, 128, 0, stream>>>(qb, kb, vb, ctxb);
    }

    // 4) output projection (fp32 out)
    {
        const int M = BB * SS, N = DD, K = DD;
        dim3 grid(N / 128, M / 128);
        gemm_bf16<float><<<grid, 256, 0, stream>>>(ctxb, wpb, bp, out, M, N, K);
    }
}
